// ExBimamba_78194174591441
// MI455X (gfx1250) — matrix operand order repaired and checked by reading, not compiled, not run on hardware
//
#include <hip/hip_runtime.h>
#include <math.h>

// ---------------------------------------------------------------------------
// Mamba block x2 for MI455X (gfx1250).
// Round 4: pure-bf16 GEMM operands. All f32->bf16 conversion is done once by
// producers (convert kernels / conv / x_proj epilogue / scan), so the WMMA
// GEMM inner loop is conversion-free: uint4 global->LDS copies + 4 v_wmma
// per wave per K-step, double-buffered, one barrier per K-step. Halves GEMM
// HBM read bytes vs f32 staging.
// ---------------------------------------------------------------------------

typedef __attribute__((ext_vector_type(16))) __bf16 v16bf;
typedef __attribute__((ext_vector_type(8)))  float  v8f;

#define D_MODEL 1024
#define D_INNER 2048
#define D_STATE 16
#define DT_RANK 64
#define BATCH   4
#define SEQLEN  1024
#define ROWS    (BATCH * SEQLEN)   // 4096 token rows

// f32 -> bf16 round-to-nearest-even (pure integer ops)
__device__ __forceinline__ unsigned f2bf(float f) {
    union { float f; unsigned u; } v; v.f = f;
    return (v.u + 0x7FFFu + ((v.u >> 16) & 1u)) >> 16;
}
__device__ __forceinline__ uint2 pack4(float4 v) {
    uint2 r;
    r.x = f2bf(v.x) | (f2bf(v.y) << 16);
    r.y = f2bf(v.z) | (f2bf(v.w) << 16);
    return r;
}

union FragBF { v16bf v; unsigned u[8]; };

// stable softplus: max(x,0) + log(1+exp(-|x|)) -> v_exp + v_log only
__device__ __forceinline__ float softplus_fast(float x) {
    return fmaxf(x, 0.f) + __logf(1.f + __expf(-fabsf(x)));
}

// ---------------------------------------------------------------------------
// Bulk f32 -> bf16 converter (vectorized float4 -> 4x bf16).
// ---------------------------------------------------------------------------
__global__ __launch_bounds__(256)
void f32_to_bf16_kernel(const float* __restrict__ src,
                        unsigned short* __restrict__ dst, int n4) {
    int i = blockIdx.x * 256 + threadIdx.x;
    if (i < n4) {
        float4 v = *(const float4*)&src[(size_t)i * 4];
        *(uint2*)&dst[(size_t)i * 4] = pack4(v);
    }
}

// ---------------------------------------------------------------------------
// Tiled GEMM, bf16 in / f32 out: Y[m,n] = sum_k A[m,k] * B[n,k]  (Y = A * B^T)
//   256 threads = 8 wave32s; macro tile 128(M) x 64(N); K-step 32.
//   Wave (wm 0..3, wn 0..1) owns 32x32 patch = 2x2 16x16 tiles -> 4 WMMA/step.
//   Double-buffered LDS + reg-staged next tile: 1 barrier per K-step.
//   Staging is pure b128 copies (operands already bf16; no conversion VALU).
//   act==1: fused softplus(acc + bias[n]) epilogue (dt_proj).
//   Ybf != null: additionally store bf16 copy of Y (x_proj -> dt_proj feed).
// Grid: (ceil(N/64), M/128). M mult of 128, K mult of 32; N guarded.
// ---------------------------------------------------------------------------
__global__ __launch_bounds__(256)
void gemm_bf16_wmma(const unsigned short* __restrict__ A, int lda,
                    const unsigned short* __restrict__ B, int ldb,
                    float* __restrict__ Y, unsigned short* __restrict__ Ybf,
                    int ldy, int N, int K,
                    const float* __restrict__ bias, int act) {
    __shared__ unsigned short As[2][128 * 40];  // 128(M) x 32(K), pad 40 shorts
    __shared__ unsigned short Bs[2][64 * 40];   //  64(N) x 32(K), pad 40 shorts

    const int tid  = threadIdx.x;
    const int lane = tid & 31;
    const int wave = tid >> 5;
    const int wm   = wave & 3;
    const int wn   = wave >> 2;
    const int mBlk = blockIdx.y * 128;
    const int nBlk = blockIdx.x * 64;

    // Staging geometry: 8-bf16 (16B) chunks; row = tid/4, chunk = tid%4.
    // LDS row stride 40 shorts = 80 B keeps every chunk 16B-aligned.
    const int rT = tid >> 2;            // 0..63
    const int cT = (tid & 3) * 8;       // K-element offset 0,8,16,24

    uint4 ra0, ra1, rb;
    auto loadTile = [&](int k0) {
        ra0 = *(const uint4*)&A[(size_t)(mBlk + rT) * lda + k0 + cT];
        ra1 = *(const uint4*)&A[(size_t)(mBlk + rT + 64) * lda + k0 + cT];
        int n = nBlk + rT;
        rb = (n < N) ? *(const uint4*)&B[(size_t)n * ldb + k0 + cT]
                     : make_uint4(0u, 0u, 0u, 0u);
    };
    auto stageTile = [&](int buf) {
        *(uint4*)&As[buf][rT * 40 + cT]        = ra0;
        *(uint4*)&As[buf][(rT + 64) * 40 + cT] = ra1;
        *(uint4*)&Bs[buf][rT * 40 + cT]        = rb;
    };

    v8f acc00 = {}, acc01 = {}, acc10 = {}, acc11 = {};

    loadTile(0);
    stageTile(0);
    __syncthreads();

    int cur = 0;
    for (int k0 = 0; k0 < K; k0 += 32) {
        const bool more = (k0 + 32) < K;
        if (more) loadTile(k0 + 32);          // prefetch next tile into regs

        // Fragments per ISA 7.12.2 bf16 layouts (row stride 40 shorts).
        const unsigned short* Ab = As[cur];
        const unsigned short* Bb = Bs[cur];
        FragBF fa[2], fb[2];
        const int kb = (lane >> 4) * 8;
#pragma unroll
        for (int mi = 0; mi < 2; ++mi) {
            const int am = (wm * 32 + mi * 16 + (lane & 15)) * 40;
#pragma unroll
            for (int i = 0; i < 4; ++i) {
                fa[mi].u[i]     = *(const unsigned*)&Ab[am + kb + 2 * i];
                fa[mi].u[4 + i] = *(const unsigned*)&Ab[am + kb + 16 + 2 * i];
            }
        }
        const int kh = (lane >> 4) * 8;
#pragma unroll
        for (int ni = 0; ni < 2; ++ni) {
            const int bn = (wn * 32 + ni * 16 + (lane & 15)) * 40;
#pragma unroll
            for (int i = 0; i < 8; ++i)
                fb[ni].u[i] = *(const unsigned*)&Bb[bn + kh + (i >> 2) * 16 + 2 * (i & 3)];
        }

        acc00 = __builtin_amdgcn_wmma_f32_16x16x32_bf16(false, fa[0].v, false, fb[0].v, (short)0, acc00, false, false);
        acc01 = __builtin_amdgcn_wmma_f32_16x16x32_bf16(false, fa[0].v, false, fb[1].v, (short)0, acc01, false, false);
        acc10 = __builtin_amdgcn_wmma_f32_16x16x32_bf16(false, fa[1].v, false, fb[0].v, (short)0, acc10, false, false);
        acc11 = __builtin_amdgcn_wmma_f32_16x16x32_bf16(false, fa[1].v, false, fb[1].v, (short)0, acc11, false, false);

        if (more) {
            stageTile(cur ^ 1);   // buf cur^1 last read before prev barrier
            __syncthreads();      // publish for next iteration
            cur ^= 1;
        }
    }

    // C/D layout: VGPR r -> M = r (lanes 0-15) or r+8 (lanes 16-31); N = lane&15
#pragma unroll
    for (int mi = 0; mi < 2; ++mi) {
#pragma unroll
        for (int ni = 0; ni < 2; ++ni) {
            const v8f a = (mi == 0) ? (ni == 0 ? acc00 : acc01)
                                    : (ni == 0 ? acc10 : acc11);
            const int n  = nBlk + wn * 32 + ni * 16 + (lane & 15);
            if (n >= N) continue;
            const int mB = mBlk + wm * 32 + mi * 16 + ((lane >> 4) << 3);
            const float bn = (act == 1) ? bias[n] : 0.f;
#pragma unroll
            for (int r = 0; r < 8; ++r) {
                float v = a[r];
                if (act == 1) v = softplus_fast(v + bn);
                Y[(size_t)(mB + r) * ldy + n] = v;
                if (Ybf) Ybf[(size_t)(mB + r) * ldy + n] = (unsigned short)f2bf(v);
            }
        }
    }
}

// ---------------------------------------------------------------------------
// Depthwise causal conv1d (width 4) + bias + SiLU. Writes f32 (scan) and
// bf16 (x_proj GEMM feed) copies of u.
// ---------------------------------------------------------------------------
__global__ __launch_bounds__(256)
void conv_silu_kernel(const float* __restrict__ xz,
                      const float* __restrict__ conv_w,
                      const float* __restrict__ conv_b,
                      float* __restrict__ u,
                      unsigned short* __restrict__ ubf) {
    const int idx = blockIdx.x * 256 + threadIdx.x;   // (b*L + l)*D_INNER + d
    const int d   = idx & (D_INNER - 1);
    const int l   = (idx >> 11) & (SEQLEN - 1);
    const int row = idx >> 11;
    float acc = conv_b[d];
#pragma unroll
    for (int j = 0; j < 4; ++j) {
        int lj = l - 3 + j;
        if (lj >= 0)
            acc += xz[(size_t)(row - 3 + j) * (2 * D_INNER) + d] * conv_w[d * 4 + j];
    }
    float s = acc / (1.f + __expf(-acc));             // silu
    u[idx]   = s;
    ubf[idx] = (unsigned short)f2bf(s);
}

// ---------------------------------------------------------------------------
// Selective scan fused with skip (u*D) and gate (*silu(z)).
// One thread per (b,d); h[16] in VGPRs; B/C chunks staged via LDS (float4).
// Output written bf16-only: its sole consumer is the out_proj WMMA GEMM.
// ---------------------------------------------------------------------------
__global__ __launch_bounds__(256)
void scan_kernel(const float* __restrict__ delta,
                 const float* __restrict__ u,
                 const float* __restrict__ xz,     // z at cols D_INNER..
                 const float* __restrict__ xdbl,   // (ROWS,96): dt|B|C (f32)
                 const float* __restrict__ A_log,
                 const float* __restrict__ D_param,
                 unsigned short* __restrict__ ybf) {
    const int b = blockIdx.x >> 3;
    const int d = ((blockIdx.x & 7) << 8) + threadIdx.x;

    float Aa[D_STATE], h[D_STATE];
#pragma unroll
    for (int n = 0; n < D_STATE; ++n) {
        Aa[n] = -__expf(A_log[d * D_STATE + n]);
        h[n]  = 0.f;
    }
    const float Dp = D_param[d];

    __shared__ float sB[64][D_STATE];
    __shared__ float sC[64][D_STATE];

    for (int l0 = 0; l0 < SEQLEN; l0 += 64) {
        __syncthreads();
#pragma unroll
        for (int i = 0; i < 2; ++i) {          // 64 rows x 8 float4 (B|C)
            int f = threadIdx.x + i * 256;
            int r = f >> 3, q = f & 7;
            float4 v = *(const float4*)&xdbl[(size_t)(b * SEQLEN + l0 + r) * 96
                                             + DT_RANK + q * 4];
            if (q < 4) *(float4*)&sB[r][q * 4] = v;
            else       *(float4*)&sC[r][(q - 4) * 4] = v;
        }
        __syncthreads();

        if (l0 + 64 < SEQLEN)
            __builtin_prefetch(&delta[(size_t)(b * SEQLEN + l0 + 64) * D_INNER + d], 0, 1);

        for (int t = 0; t < 64; ++t) {
            const size_t row = (size_t)(b * SEQLEN + l0 + t);
            const float dl = delta[row * D_INNER + d];
            const float uv = u[row * D_INNER + d];
            const float zv = xz[row * (2 * D_INNER) + D_INNER + d];
            const float du = dl * uv;
            float y = 0.f;
#pragma unroll
            for (int n = 0; n < D_STATE; ++n) {
                h[n] = h[n] * __expf(dl * Aa[n]) + du * sB[t][n];
                y   += h[n] * sC[t][n];
            }
            y += uv * Dp;                        // skip connection
            y *= zv / (1.f + __expf(-zv));       // * silu(z)
            ybf[row * D_INNER + d] = (unsigned short)f2bf(y);
        }
    }
}

// ---------------------------------------------------------------------------
// Orchestration. Workspace reused for both blocks (~183 MiB):
//   f32:  xz 64MB | u 32MB | xdbl 1.5MB | delta 32MB
//   bf16: x 8MB | W_in 8MB | W_xp 0.4MB | W_dt 0.25MB | W_out 4MB
//         u 16MB | xdbl 0.75MB | y 16MB
// ---------------------------------------------------------------------------
extern "C" void kernel_launch(void* const* d_in, const int* in_sizes, int n_in,
                              void* d_out, int out_size, void* d_ws, size_t ws_size,
                              hipStream_t stream) {
    (void)in_sizes; (void)n_in; (void)out_size; (void)ws_size;

    float* out = (float*)d_out;
    float* ws  = (float*)d_ws;
    // f32 region
    float* xz   = ws;
    float* ubuf = xz   + (size_t)ROWS * (2 * D_INNER);
    float* xdbl = ubuf + (size_t)ROWS * D_INNER;
    float* dlt  = xdbl + (size_t)ROWS * 96;
    // bf16 region (all sizes keep 16B alignment)
    unsigned short* bf = (unsigned short*)(dlt + (size_t)ROWS * D_INNER);
    unsigned short* xbf     = bf;
    unsigned short* winbf   = xbf     + (size_t)ROWS * D_MODEL;
    unsigned short* wxpbf   = winbf   + (size_t)(2 * D_INNER) * D_MODEL;
    unsigned short* wdtbf   = wxpbf   + (size_t)96 * D_INNER;
    unsigned short* woutbf  = wdtbf   + (size_t)D_INNER * DT_RANK;
    unsigned short* ubf     = woutbf  + (size_t)D_MODEL * D_INNER;
    unsigned short* xdblbf  = ubf     + (size_t)ROWS * D_INNER;
    unsigned short* ybf     = xdblbf  + (size_t)ROWS * 96;

    for (int blk = 0; blk < 2; ++blk) {
        const float* x          = (const float*)d_in[blk];            // hidden/diff
        const float* in_proj_w  = (const float*)d_in[2 + 9 * blk + 0];
        const float* conv_w     = (const float*)d_in[2 + 9 * blk + 1];
        const float* conv_b     = (const float*)d_in[2 + 9 * blk + 2];
        const float* x_proj_w   = (const float*)d_in[2 + 9 * blk + 3];
        const float* dt_proj_w  = (const float*)d_in[2 + 9 * blk + 4];
        const float* dt_proj_b  = (const float*)d_in[2 + 9 * blk + 5];
        const float* A_log      = (const float*)d_in[2 + 9 * blk + 6];
        const float* D_param    = (const float*)d_in[2 + 9 * blk + 7];
        const float* out_proj_w = (const float*)d_in[2 + 9 * blk + 8];
        float* yout = out + (size_t)blk * ROWS * D_MODEL;

        // 0) one-time bf16 mirrors of input + weights for this block
        auto cvt = [&](const float* s, unsigned short* dsts, size_t n) {
            int n4 = (int)(n / 4);
            f32_to_bf16_kernel<<<(n4 + 255) / 256, 256, 0, stream>>>(s, dsts, n4);
        };
        cvt(x,          xbf,    (size_t)ROWS * D_MODEL);
        cvt(in_proj_w,  winbf,  (size_t)(2 * D_INNER) * D_MODEL);
        cvt(x_proj_w,   wxpbf,  (size_t)96 * D_INNER);
        cvt(dt_proj_w,  wdtbf,  (size_t)D_INNER * DT_RANK);
        cvt(out_proj_w, woutbf, (size_t)D_MODEL * D_INNER);

        // 1) xz = x @ in_proj_w^T          (4096 x 4096, K=1024)
        gemm_bf16_wmma<<<dim3((2 * D_INNER) / 64, ROWS / 128), 256, 0, stream>>>(
            xbf, D_MODEL, winbf, D_MODEL, xz, nullptr, 2 * D_INNER,
            2 * D_INNER, D_MODEL, nullptr, 0);

        // 2) u = silu(causal depthwise conv(x_in) + b)   (f32 + bf16 copies)
        conv_silu_kernel<<<(ROWS * D_INNER) / 256, 256, 0, stream>>>(
            xz, conv_w, conv_b, ubuf, ubf);

        // 3) x_dbl = u @ x_proj_w^T        (4096 x 96, K=2048; N guarded)
        //    writes f32 (scan B/C) + bf16 (dt_proj A operand)
        gemm_bf16_wmma<<<dim3((96 + 63) / 64, ROWS / 128), 256, 0, stream>>>(
            ubf, D_INNER, wxpbf, D_INNER, xdbl, xdblbf, 96,
            96, D_INNER, nullptr, 0);

        // 4) delta = softplus(dt @ dt_proj_w^T + b)   (4096 x 2048, K=64)
        gemm_bf16_wmma<<<dim3(D_INNER / 64, ROWS / 128), 256, 0, stream>>>(
            xdblbf, 96, wdtbf, DT_RANK, dlt, nullptr, D_INNER,
            D_INNER, DT_RANK, dt_proj_b, 1);

        // 5) selective scan + skip + gate  -> bf16 y (feeds out_proj only)
        scan_kernel<<<(BATCH * D_INNER) / 256, 256, 0, stream>>>(
            dlt, ubuf, xz, xdbl, A_log, D_param, ybf);

        // 6) out = y @ out_proj_w^T        (4096 x 1024, K=2048)
        gemm_bf16_wmma<<<dim3(D_MODEL / 64, ROWS / 128), 256, 0, stream>>>(
            ybf, D_INNER, woutbf, D_INNER, yout, nullptr, D_MODEL,
            D_MODEL, D_INNER, nullptr, 0);
    }
}
